// SentimentModel_28321014349947
// MI455X (gfx1250) — matrix operand order repaired and checked by reading, not compiled, not run on hardware
//
#include <hip/hip_runtime.h>
#include <hip/hip_bf16.h>

typedef __attribute__((ext_vector_type(16))) _Float16 v16h;
typedef __attribute__((ext_vector_type(8)))  _Float16 v8h;
typedef __attribute__((ext_vector_type(8)))  float    v8f;
typedef __attribute__((ext_vector_type(4)))  float    v4f;

#define BATCH  128
#define SEQT   512
#define EMBED  128
#define HIDDEN 256

// tiling: 8 batch groups x 8 hidden slices = 64 workgroups
#define MB     16    // batch rows per group (one M tile)
#define NSLICE 8     // hidden slices per batch group
#define JW     32    // hidden units per WG
#define NW     128   // gate columns per WG = 4*JW  (i|f|g|o blocks of 32)
#define GS     132   // gates LDS row stride (f32, padded)

static __device__ __forceinline__ float sigf(float x) {
  return 1.0f / (1.0f + __expf(-x));
}

static __device__ __forceinline__ v16h pack16(v8h lo, v8h hi) {
  v16h r;
#pragma unroll
  for (int i = 0; i < 8; ++i) { r[i] = lo[i]; r[i + 8] = hi[i]; }
  return r;
}

__global__ void lstm_init_kernel(int* ctr) {
  if (threadIdx.x < 16) ctr[threadIdx.x] = 0;
}

// One persistent kernel per LSTM layer. 64 WGs x 256 threads (8 wave32).
// WG = (batch group bg, hidden slice s). Owns gate rows {g*256 + s*32 .. +32}
// for g in {0..3} (i,f,g,o), holds its f16 weight slice in LDS for all T steps.
template <int KIN, bool IS_L0>
__global__ __launch_bounds__(256) void lstm_layer_kernel(
    const int*   __restrict__ x,      // (B,T) token ids (layer0 only)
    const float* __restrict__ emb,    // (VOCAB, EMBED)  (layer0 only)
    const float* __restrict__ W_ih,   // (1024, D)
    const float* __restrict__ W_hh,   // (1024, 256)
    const float* __restrict__ b_ih,   // (1024)
    const float* __restrict__ b_hh,   // (1024)
    _Float16*    __restrict__ H0,     // (B,T,256) f16: layer0 writes, layer1 reads
    _Float16*    __restrict__ hb0,    // (B,256) f16 ping
    _Float16*    __restrict__ hb1,    // (B,256) f16 pong
    int*         __restrict__ ctr_base)
{
  constexpr int D  = KIN - HIDDEN;  // x-feature width (128 or 256)
  constexpr int AS = KIN + 8;       // A row stride in f16 (16B-aligned rows, bank-spread)
  constexpr int WS = KIN + 8;       // weight row stride in f16

  extern __shared__ char smem[];
  _Float16* wlds = (_Float16*)smem;           // NW x WS   weight slice (f16)
  _Float16* alds = wlds + NW * WS;            // MB x AS   A = [x_t | h_{t-1}]
  float*    glds = (float*)(alds + MB * AS);  // MB x GS   pre-activation gates
  float*    blds = glds + MB * GS;            // NW        fused bias b_ih+b_hh

  const int tid   = threadIdx.x;
  const int lane  = tid & 31;
  const int wave  = tid >> 5;                 // 0..7 -> N tile
  const int slice = blockIdx.x & (NSLICE - 1);
  const int bg    = blockIdx.x >> 3;          // 0..7
  const int j0    = slice * JW;               // first owned hidden unit
  int* ctr = ctr_base + bg;

  // ---- one-time: stage weight slice (f32 -> f16) and fused bias into LDS
  for (int idx = tid; idx < NW * (KIN / 4); idx += 256) {
    const int r    = idx / (KIN / 4);
    const int cc   = (idx % (KIN / 4)) * 4;
    const int grow = (r >> 5) * HIDDEN + j0 + (r & 31);  // global gate row
    v4f w;
    if (cc < D) w = *(const v4f*)(W_ih + grow * D + cc);
    else        w = *(const v4f*)(W_hh + grow * HIDDEN + (cc - D));
    _Float16* dst = wlds + r * WS + cc;
#pragma unroll
    for (int i = 0; i < 4; ++i) dst[i] = (_Float16)w[i];
  }
  if (tid < NW) {
    const int grow = (tid >> 5) * HIDDEN + j0 + (tid & 31);
    blds[tid] = b_ih[grow] + b_hh[grow];
  }

  float creg[2] = {0.0f, 0.0f};  // persistent cell state: 2 (m,j) pairs per thread

  const int am  = tid >> 4;  // 0..15 A row
  const int seg = tid & 15;  // 0..15 column segment

  for (int t = 0; t < SEQT; ++t) {
    const _Float16* hprev = (t & 1) ? hb0 : hb1;  // parity (t-1)&1

    // ---- stage A = [x_t | h_{t-1}] as f16
    if (IS_L0) {
      const int tok = x[(bg * MB + am) * SEQT + t];
      const float* e = emb + (long)tok * EMBED + seg * 8;
      v4f e0 = *(const v4f*)(e);
      v4f e1 = *(const v4f*)(e + 4);
      _Float16* dst = alds + am * AS + seg * 8;
#pragma unroll
      for (int i = 0; i < 4; ++i) { dst[i] = (_Float16)e0[i]; dst[4 + i] = (_Float16)e1[i]; }
    } else {
      const _Float16* src = H0 + ((long)(bg * MB + am) * SEQT + t) * HIDDEN + seg * 16;
      _Float16* dst = alds + am * AS + seg * 16;
      *(v8h*)dst       = *(const v8h*)src;
      *(v8h*)(dst + 8) = *(const v8h*)(src + 8);
      if (t + 1 < SEQT)
        __builtin_prefetch(src + HIDDEN, 0, 3);  // next step's x row
    }
    {
      _Float16* dst = alds + am * AS + D + seg * 16;
      if (t == 0) {
        v8h z = {};
        *(v8h*)dst = z; *(v8h*)(dst + 8) = z;
      } else {
        const _Float16* src = hprev + (bg * MB + am) * HIDDEN + seg * 16;
        *(v8h*)dst       = *(const v8h*)src;
        *(v8h*)(dst + 8) = *(const v8h*)(src + 8);
      }
    }
    __syncthreads();

    // ---- GEMM: each wave computes one 16x16 gate tile with f16 WMMA (f32 acc)
    {
      v8f acc = {};
      const int row   = lane & 15;
      const int koffA = (lane >> 4) * 16;  // A: 16 contiguous K per lane
      const int koffB = (lane >> 4) * 16;  // B: 16 contiguous K per lane
      const _Float16* arow = alds + row * AS + koffA;
      const _Float16* wrow = wlds + (wave * 16 + row) * WS + koffB;
#pragma unroll
      for (int kt = 0; kt < KIN / 32; ++kt) {
        const _Float16* ap = arow + kt * 32;
        const _Float16* bp = wrow + kt * 32;
        v16h afrag = pack16(*(const v8h*)ap, *(const v8h*)(ap + 8));
        v16h bfrag = pack16(*(const v8h*)bp, *(const v8h*)(bp + 8));
        acc = __builtin_amdgcn_wmma_f32_16x16x32_f16(false, afrag, false, bfrag,
                                                     (short)0, acc, false, false);
      }
      // C layout: VGPR v -> M = v + 8*(lane/16), N = lane%16
      const int cn    = wave * 16 + (lane & 15);
      const int rbase = (lane >> 4) * 8;
#pragma unroll
      for (int v = 0; v < 8; ++v)
        glds[(rbase + v) * GS + cn] = acc[v];
    }
    __syncthreads();

    // ---- elementwise LSTM cell update (c lives in registers across all t)
#pragma unroll
    for (int q = 0; q < 2; ++q) {
      const int p = tid + q * 256;   // 512 (m,j) pairs
      const int j = p & 31;
      const int m = (p >> 5) & 15;
      const float gi = sigf (glds[m * GS +      j] + blds[     j]);
      const float gf = sigf (glds[m * GS + 32 + j] + blds[32 + j]);
      const float gg = tanhf(glds[m * GS + 64 + j] + blds[64 + j]);
      const float go = sigf (glds[m * GS + 96 + j] + blds[96 + j]);
      const float c  = gf * creg[q] + gi * gg;
      creg[q] = c;
      const float h  = go * tanhf(c);
      const _Float16 hh = (_Float16)h;
      const int bglob = bg * MB + m;
      const int jglob = j0 + j;
      ((t & 1) ? hb1 : hb0)[bglob * HIDDEN + jglob] = hh;
      if (IS_L0) H0[((long)bglob * SEQT + t) * HIDDEN + jglob] = hh;
    }

    // ---- barrier among the 8 WGs of this batch group (monotonic counter)
    __threadfence();        // make this wave's h stores device-visible
    __syncthreads();
    if (tid == 0) {
      atomicAdd(ctr, 1);
      const int target = NSLICE * (t + 1);
      while (__hip_atomic_load(ctr, __ATOMIC_RELAXED, __HIP_MEMORY_SCOPE_AGENT) < target)
        __builtin_amdgcn_s_sleep(1);
      __threadfence();      // acquire: invalidate for peer h reads
    }
    __syncthreads();
  }
}

__global__ void fc_kernel(const _Float16* __restrict__ hlast,  // (B,256) f16
                          const float* __restrict__ fc_w,      // (256)
                          const float* __restrict__ fc_b,      // (1)
                          float* __restrict__ out)             // (B,1)
{
  const int b = threadIdx.x;
  if (b < BATCH) {
    float s = fc_b[0];
#pragma unroll 8
    for (int k = 0; k < HIDDEN; ++k)
      s += (float)hlast[b * HIDDEN + k] * fc_w[k];
    out[b] = 1.0f / (1.0f + __expf(-s));
  }
}

extern "C" void kernel_launch(void* const* d_in, const int* in_sizes, int n_in,
                              void* d_out, int out_size, void* d_ws, size_t ws_size,
                              hipStream_t stream)
{
  (void)in_sizes; (void)n_in; (void)out_size; (void)ws_size;
  const int*   x     = (const int*)  d_in[0];
  const float* emb   = (const float*)d_in[1];
  const float* fc_w  = (const float*)d_in[2];
  const float* fc_b  = (const float*)d_in[3];
  const float* W_ih0 = (const float*)d_in[4];
  const float* W_hh0 = (const float*)d_in[5];
  const float* b_ih0 = (const float*)d_in[6];
  const float* b_hh0 = (const float*)d_in[7];
  const float* W_ih1 = (const float*)d_in[8];
  const float* W_hh1 = (const float*)d_in[9];
  const float* b_ih1 = (const float*)d_in[10];
  const float* b_hh1 = (const float*)d_in[11];

  // workspace layout
  char*     ws  = (char*)d_ws;
  int*      ctr = (int*)ws;                      // 16 counters (8 per layer)
  _Float16* hb0 = (_Float16*)(ws + 256);         // (B,256) ping
  _Float16* hb1 = hb0 + BATCH * HIDDEN;          // (B,256) pong
  _Float16* H0  = hb1 + BATCH * HIDDEN;          // (B,T,256) layer0 output, ~32MB

  lstm_init_kernel<<<1, 64, 0, stream>>>(ctr);

  constexpr int K0 = EMBED  + HIDDEN;  // 384
  constexpr int K1 = HIDDEN + HIDDEN;  // 512
  const size_t smem0 = (size_t)NW * (K0 + 8) * 2 + (size_t)MB * (K0 + 8) * 2
                     + (size_t)MB * GS * 4 + NW * 4;   // ~122 KB
  const size_t smem1 = (size_t)NW * (K1 + 8) * 2 + (size_t)MB * (K1 + 8) * 2
                     + (size_t)MB * GS * 4 + NW * 4;   // ~159 KB (CDNA5: 320KB/WG LDS)

  lstm_layer_kernel<K0, true ><<<64, 256, smem0, stream>>>(
      x, emb, W_ih0, W_hh0, b_ih0, b_hh0, H0, hb0, hb1, ctr);
  lstm_layer_kernel<K1, false><<<64, 256, smem1, stream>>>(
      x, emb, W_ih1, W_hh1, b_ih1, b_hh1, H0, hb0, hb1, ctr + 8);

  // last timestep t = 511 has parity 1 -> h_T lives in hb1
  fc_kernel<<<1, 128, 0, stream>>>(hb1, fc_w, fc_b, (float*)d_out);
}
